// SparseLayer_48876727828533
// MI455X (gfx1250) — matrix operand order repaired and checked by reading, not compiled, not run on hardware
//
#include <hip/hip_runtime.h>

typedef __attribute__((ext_vector_type(16))) __bf16 v16bf;
typedef __attribute__((ext_vector_type(8)))  __bf16 v8bf;
typedef __attribute__((ext_vector_type(4)))  __bf16 v4bf;
typedef __attribute__((ext_vector_type(8)))  float  v8f;
typedef __attribute__((ext_vector_type(4)))  float  v4f;
typedef __attribute__((ext_vector_type(4)))  unsigned int u32x4;
typedef __attribute__((ext_vector_type(8)))  int   i32x8;
typedef __attribute__((ext_vector_type(4)))  int   i32x4;

#define SIZE_K 2048   // size_in
#define SIZE_M 2048   // size_out (rows of out)
#define SIZE_N 8192   // batch    (cols of out)
#define BM 128
#define BN 128
#define BK 32
#define PLANE (BM * BK)           // bf16 elements per LDS plane (4096)
#define PLANE_BYTES (PLANE * 2)   // 8 KB
#define NK (SIZE_K / BK)          // 64 K-steps

union Frag {
    v16bf v;
    struct { v8bf lo, hi; } p;
};

// ---------------------------------------------------------------------------
// TDM: 2-D tile load Global -> LDS.  tile = rows x cols (cols contiguous,
// 2-byte elements), row stride = stride0 elements, tensor extents td0 x td1.
// D# packing per CDNA5 ISA 8.3/8.4 (group0: count/lds/global/type,
// group1: data_size, dims, tile dims, dim0 stride; groups 2-3 zero for 2-D).
// 6-arg builtin form (clang-23 / therock-10.0 headers).
// ---------------------------------------------------------------------------
__device__ inline void tdm_tile_load(unsigned lds_off, const __bf16* gptr,
                                     unsigned rows, unsigned td1,
                                     unsigned cols, unsigned td0,
                                     unsigned stride0)
{
#if defined(__HIP_DEVICE_COMPILE__)
    unsigned long long ga = (unsigned long long)(uintptr_t)gptr;
    u32x4 g0;
    g0[0] = 1u;                                                   // count=1, user mode
    g0[1] = lds_off;                                              // LDS byte address
    g0[2] = (unsigned)ga;                                         // global_addr[31:0]
    g0[3] = ((unsigned)(ga >> 32) & 0x1FFFFFFu) | (2u << 30);     // addr[56:32] | type=2
    i32x8 g1;
    g1[0] = (int)(1u << 16);                                      // data_size=1 (2 bytes)
    g1[1] = (int)((td0 & 0xFFFFu) << 16);                         // tensor_dim0[15:0]
    g1[2] = (int)((td0 >> 16) | ((td1 & 0xFFFFu) << 16));         // td0[31:16] | td1[15:0]
    g1[3] = (int)((td1 >> 16) | ((cols & 0xFFFFu) << 16));        // td1[31:16] | tile_dim0
    g1[4] = (int)(rows & 0xFFFFu);                                // tile_dim1 (tile_dim2=0)
    g1[5] = (int)stride0;                                         // tensor_dim0_stride[31:0]
    g1[6] = 0;                                                    // stride0[47:32] | stride1 lo
    g1[7] = 0;
    i32x4 z4 = {0, 0, 0, 0};
    i32x8 z8 = {0, 0, 0, 0, 0, 0, 0, 0};
    __builtin_amdgcn_tensor_load_to_lds(g0, g1, z4, z4, z8, 0);
#else
    (void)lds_off; (void)gptr; (void)rows; (void)td1; (void)cols; (void)td0; (void)stride0;
#endif
}

// ---------------------------------------------------------------------------
// Pre-pass 1: split X (fp32 [N,K]) into bf16 hi/lo planes, same layout.
// ---------------------------------------------------------------------------
__global__ __launch_bounds__(256)
void convert_split_x(const float* __restrict__ X, __bf16* __restrict__ Xh,
                     __bf16* __restrict__ Xl)
{
    size_t i = ((size_t)blockIdx.x * 256 + threadIdx.x) * 4;
    v4f v = *(const v4f*)(X + i);
    v4bf h, l;
    #pragma unroll
    for (int c = 0; c < 4; ++c) {
        h[c] = (__bf16)v[c];
        l[c] = (__bf16)(v[c] - (float)h[c]);
    }
    *(v4bf*)(Xh + i) = h;
    *(v4bf*)(Xl + i) = l;
}

// ---------------------------------------------------------------------------
// Pre-pass 2: W [K,M] fp32 -> transposed bf16 hi/lo planes Wt [M,K].
// LDS-tiled 32x32 transpose, coalesced loads and stores.
// ---------------------------------------------------------------------------
__global__ __launch_bounds__(256)
void convert_w_transpose(const float* __restrict__ W, __bf16* __restrict__ Wth,
                         __bf16* __restrict__ Wtl)
{
    __shared__ float tile[32][33];
    const int c  = threadIdx.x & 31;
    const int r4 = (threadIdx.x >> 5) * 4;
    const int obase = blockIdx.x * 32;
    const int kbase = blockIdx.y * 32;
    #pragma unroll
    for (int i = 0; i < 4; ++i)
        tile[r4 + i][c] = W[(size_t)(kbase + r4 + i) * SIZE_M + obase + c];
    __syncthreads();
    #pragma unroll
    for (int i = 0; i < 4; ++i) {
        float f = tile[c][r4 + i];                 // = W[kbase+c][obase+r4+i]
        __bf16 h = (__bf16)f;
        __bf16 l = (__bf16)(f - (float)h);
        size_t o = (size_t)(obase + r4 + i) * SIZE_K + kbase + c;
        Wth[o] = h;
        Wtl[o] = l;
    }
}

// ---------------------------------------------------------------------------
// Main GEMM: out[o,b] = sum_k Wt[o,k] * X[b,k], bf16x3 split accumulation,
// double-buffered LDS fed by the Tensor Data Mover.
// ---------------------------------------------------------------------------
__global__ __launch_bounds__(256)
void gemm_bf16x3_tdm(const __bf16* __restrict__ Wth, const __bf16* __restrict__ Wtl,
                     const __bf16* __restrict__ Xh, const __bf16* __restrict__ Xl,
                     float* __restrict__ Out, int zero_guard)
{
    __shared__ __bf16 sm[2 * 4 * PLANE];   // [buf][Ah|Al|Bh|Bl][128x32], 64 KB
    if (zero_guard) sm[threadIdx.x] = (__bf16)0.0f;   // never taken; keeps LDS "written"

    const int tid    = threadIdx.x;
    const int lane   = tid & 31;
    const int wave   = tid >> 5;
    const int laneM  = lane & 15;
    const int laneHi = lane >> 4;
    const int wm     = wave >> 2;    // 0..1 -> 64-row M slab
    const int wn     = wave & 3;     // 0..3 -> 32-col N slab

    const int oBase = blockIdx.y * BM;
    const int bBase = blockIdx.x * BN;

    const unsigned smBase = (unsigned)(uintptr_t)&sm[0];

    v8f acc[4][2] = {};

    // stage(s): issue the 4 plane loads for K-step s into buffer s&1
    auto stage = [&](int s) {
        const int k0 = s * BK;
        const unsigned base = smBase + (unsigned)((s & 1) * 4) * PLANE_BYTES;
        tdm_tile_load(base + 0 * PLANE_BYTES, Wth + (size_t)oBase * SIZE_K + k0,
                      BM, SIZE_M, BK, SIZE_K, SIZE_K);
        tdm_tile_load(base + 1 * PLANE_BYTES, Wtl + (size_t)oBase * SIZE_K + k0,
                      BM, SIZE_M, BK, SIZE_K, SIZE_K);
        tdm_tile_load(base + 2 * PLANE_BYTES, Xh + (size_t)bBase * SIZE_K + k0,
                      BN, SIZE_N, BK, SIZE_K, SIZE_K);
        tdm_tile_load(base + 3 * PLANE_BYTES, Xl + (size_t)bBase * SIZE_K + k0,
                      BN, SIZE_N, BK, SIZE_K, SIZE_K);
    };

    if (wave == 0) stage(0);

    for (int kt = 0; kt < NK; ++kt) {
        const int buf = kt & 1;
        if (wave == 0) {
            if (kt + 1 < NK) {
                stage(kt + 1);                              // prefetch next buffer
                __builtin_amdgcn_s_wait_tensorcnt(4);       // this buffer's 4 loads done
            } else {
                __builtin_amdgcn_s_wait_tensorcnt(0);
            }
        }
        __syncthreads();   // data visible to all waves

        const __bf16* Ah = sm + (size_t)(buf * 4 + 0) * PLANE;
        const __bf16* Al = sm + (size_t)(buf * 4 + 1) * PLANE;
        const __bf16* Bh = sm + (size_t)(buf * 4 + 2) * PLANE;
        const __bf16* Bl = sm + (size_t)(buf * 4 + 3) * PLANE;

        Frag a_h[4], a_l[4], b_hf[2], b_lf[2];
        #pragma unroll
        for (int i = 0; i < 4; ++i) {
            const int o = wm * 64 + i * 16 + laneM;
            a_h[i].p.lo = *(const v8bf*)(Ah + o * BK + laneHi * 8);
            a_h[i].p.hi = *(const v8bf*)(Ah + o * BK + 16 + laneHi * 8);
            a_l[i].p.lo = *(const v8bf*)(Al + o * BK + laneHi * 8);
            a_l[i].p.hi = *(const v8bf*)(Al + o * BK + 16 + laneHi * 8);
        }
        #pragma unroll
        for (int j = 0; j < 2; ++j) {
            const int b = wn * 32 + j * 16 + laneM;
            b_hf[j].p.lo = *(const v8bf*)(Bh + b * BK + laneHi * 8);
            b_hf[j].p.hi = *(const v8bf*)(Bh + b * BK + 16 + laneHi * 8);
            b_lf[j].p.lo = *(const v8bf*)(Bl + b * BK + laneHi * 8);
            b_lf[j].p.hi = *(const v8bf*)(Bl + b * BK + 16 + laneHi * 8);
        }

        #pragma unroll
        for (int i = 0; i < 4; ++i)
            #pragma unroll
            for (int j = 0; j < 2; ++j) {
                acc[i][j] = __builtin_amdgcn_wmma_f32_16x16x32_bf16(
                    false, a_h[i].v, false, b_hf[j].v, (short)0, acc[i][j], false, false);
                acc[i][j] = __builtin_amdgcn_wmma_f32_16x16x32_bf16(
                    false, a_l[i].v, false, b_hf[j].v, (short)0, acc[i][j], false, false);
                acc[i][j] = __builtin_amdgcn_wmma_f32_16x16x32_bf16(
                    false, a_h[i].v, false, b_lf[j].v, (short)0, acc[i][j], false, false);
            }

        __syncthreads();   // compute done -> buffer may be overwritten
    }

    #pragma unroll
    for (int i = 0; i < 4; ++i)
        #pragma unroll
        for (int j = 0; j < 2; ++j) {
            const int o0 = oBase + wm * 64 + i * 16 + laneHi * 8;
            const int b  = bBase + wn * 32 + j * 16 + laneM;
            #pragma unroll
            for (int r = 0; r < 8; ++r)
                Out[(size_t)(o0 + r) * SIZE_N + b] = acc[i][j][r];
        }
}

// ---------------------------------------------------------------------------
// Fallback fused kernel (used only if the workspace is too small): converts
// fp32->bf16 hi/lo per tile inside the GEMM (round-1 version).
// ---------------------------------------------------------------------------
__global__ __launch_bounds__(256)
void sparse_layer_gemm_fused(const float* __restrict__ X,
                             const float* __restrict__ W,
                             float* __restrict__ Out)
{
    __shared__ __bf16 lds[4 * PLANE];
    __bf16* Ah = lds;
    __bf16* Al = lds + 1 * PLANE;
    __bf16* Bh = lds + 2 * PLANE;
    __bf16* Bl = lds + 3 * PLANE;

    const int tid    = threadIdx.x;
    const int lane   = tid & 31;
    const int wave   = tid >> 5;
    const int laneM  = lane & 15;
    const int laneHi = lane >> 4;
    const int wm     = wave >> 2;
    const int wn     = wave & 3;

    const int oBase = blockIdx.y * BM;
    const int bBase = blockIdx.x * BN;

    const int a_k  = tid >> 3;
    const int a_oo = (tid & 7) * 16;
    const int b_b  = tid >> 1;
    const int b_ko = (tid & 1) * 16;

    v8f acc[4][2] = {};

    for (int kt = 0; kt < NK; ++kt) {
        const int k0 = kt * BK;
        __syncthreads();
        {
            const v4f* wp = (const v4f*)(W + (size_t)(k0 + a_k) * SIZE_M + oBase + a_oo);
            float f[16];
            #pragma unroll
            for (int q = 0; q < 4; ++q) {
                v4f v = wp[q];
                #pragma unroll
                for (int c = 0; c < 4; ++c) f[q * 4 + c] = v[c];
            }
            #pragma unroll
            for (int e = 0; e < 16; ++e) {
                const int o = a_oo + e;
                __bf16 h = (__bf16)f[e];
                __bf16 l = (__bf16)(f[e] - (float)h);
                Ah[o * BK + a_k] = h;
                Al[o * BK + a_k] = l;
            }
        }
        {
            const v4f* xp = (const v4f*)(X + (size_t)(bBase + b_b) * SIZE_K + k0 + b_ko);
            v8bf hb0, hb1, lb0, lb1;
            #pragma unroll
            for (int q = 0; q < 4; ++q) {
                v4f v = xp[q];
                #pragma unroll
                for (int c = 0; c < 4; ++c) {
                    const int e = q * 4 + c;
                    __bf16 h = (__bf16)v[c];
                    __bf16 l = (__bf16)(v[c] - (float)h);
                    if (e < 8) { hb0[e] = h;     lb0[e] = l; }
                    else       { hb1[e - 8] = h; lb1[e - 8] = l; }
                }
            }
            *(v8bf*)(Bh + b_b * BK + b_ko)     = hb0;
            *(v8bf*)(Bh + b_b * BK + b_ko + 8) = hb1;
            *(v8bf*)(Bl + b_b * BK + b_ko)     = lb0;
            *(v8bf*)(Bl + b_b * BK + b_ko + 8) = lb1;
        }
        __syncthreads();

        Frag a_h[4], a_l[4], b_hf[2], b_lf[2];
        #pragma unroll
        for (int i = 0; i < 4; ++i) {
            const int o = wm * 64 + i * 16 + laneM;
            a_h[i].p.lo = *(const v8bf*)(Ah + o * BK + laneHi * 8);
            a_h[i].p.hi = *(const v8bf*)(Ah + o * BK + 16 + laneHi * 8);
            a_l[i].p.lo = *(const v8bf*)(Al + o * BK + laneHi * 8);
            a_l[i].p.hi = *(const v8bf*)(Al + o * BK + 16 + laneHi * 8);
        }
        #pragma unroll
        for (int j = 0; j < 2; ++j) {
            const int b = wn * 32 + j * 16 + laneM;
            b_hf[j].p.lo = *(const v8bf*)(Bh + b * BK + laneHi * 8);
            b_hf[j].p.hi = *(const v8bf*)(Bh + b * BK + 16 + laneHi * 8);
            b_lf[j].p.lo = *(const v8bf*)(Bl + b * BK + laneHi * 8);
            b_lf[j].p.hi = *(const v8bf*)(Bl + b * BK + 16 + laneHi * 8);
        }
        #pragma unroll
        for (int i = 0; i < 4; ++i)
            #pragma unroll
            for (int j = 0; j < 2; ++j) {
                acc[i][j] = __builtin_amdgcn_wmma_f32_16x16x32_bf16(
                    false, a_h[i].v, false, b_hf[j].v, (short)0, acc[i][j], false, false);
                acc[i][j] = __builtin_amdgcn_wmma_f32_16x16x32_bf16(
                    false, a_l[i].v, false, b_hf[j].v, (short)0, acc[i][j], false, false);
                acc[i][j] = __builtin_amdgcn_wmma_f32_16x16x32_bf16(
                    false, a_h[i].v, false, b_lf[j].v, (short)0, acc[i][j], false, false);
            }
    }

    #pragma unroll
    for (int i = 0; i < 4; ++i)
        #pragma unroll
        for (int j = 0; j < 2; ++j) {
            const int o0 = oBase + wm * 64 + i * 16 + laneHi * 8;
            const int b  = bBase + wn * 32 + j * 16 + laneM;
            #pragma unroll
            for (int r = 0; r < 8; ++r)
                Out[(size_t)(o0 + r) * SIZE_N + b] = acc[i][j][r];
        }
}

extern "C" void kernel_launch(void* const* d_in, const int* in_sizes, int n_in,
                              void* d_out, int out_size, void* d_ws, size_t ws_size,
                              hipStream_t stream) {
    const float* X = (const float*)d_in[0];   // [8192, 2048]
    const float* W = (const float*)d_in[1];   // [2048, 2048]
    // d_in[2] = bias: reference discards the bias add, so it is unused.
    float* out = (float*)d_out;               // [2048, 8192]

    const size_t wEl = (size_t)SIZE_K * SIZE_M;          // 4,194,304
    const size_t xEl = (size_t)SIZE_N * SIZE_K;          // 16,777,216
    const size_t need = (2 * wEl + 2 * xEl) * sizeof(__bf16);   // 80 MB

    if (ws_size >= need) {
        __bf16* Wth = (__bf16*)d_ws;
        __bf16* Wtl = Wth + wEl;
        __bf16* Xh  = Wtl + wEl;
        __bf16* Xl  = Xh + xEl;
        convert_split_x<<<(unsigned)(xEl / (256 * 4)), 256, 0, stream>>>(X, Xh, Xl);
        convert_w_transpose<<<dim3(SIZE_M / 32, SIZE_K / 32), 256, 0, stream>>>(W, Wth, Wtl);
        gemm_bf16x3_tdm<<<dim3(SIZE_N / BN, SIZE_M / BM), 256, 0, stream>>>(
            Wth, Wtl, Xh, Xl, out, 0);
    } else {
        sparse_layer_gemm_fused<<<dim3(SIZE_N / BN, SIZE_M / BM), 256, 0, stream>>>(X, W, out);
    }
}
